// ADRC_PE_63247688401324
// MI455X (gfx1250) — matrix operand order repaired and checked by reading, not compiled, not run on hardware
//
#include <hip/hip_runtime.h>
#include <hip/hip_bf16.h>
#include <math.h>

typedef __attribute__((ext_vector_type(16))) __bf16 v16bf;
typedef __attribute__((ext_vector_type(8)))  float  v8f;

typedef __attribute__((address_space(1))) int g_i32;   // global
typedef __attribute__((address_space(3))) int l_i32;   // LDS

#define BB     8
#define CIN    256
#define CREDN  64
#define HH     160
#define WWD    160
#define HWP    (HH * WWD)            /* 25600 */
#define NYELT  (BB * CREDN * HWP)    /* 13,107,200 floats = 52.4 MB */
#define PBLK   128                   /* pixels per block in pass 1 */
#define NBLK1  (BB * (HWP / PBLK))   /* 1600 */

#if __has_builtin(__builtin_amdgcn_global_load_async_to_lds_b32) && \
    __has_builtin(__builtin_amdgcn_s_wait_asynccnt)
#define USE_ASYNC_LDS 1
#else
#define USE_ASYNC_LDS 0
#endif

// ---------------------------------------------------------------------------
// Pass 1: y[b,o,p] = sum_c W[o,c] * x[b,c,p]  via bf16 WMMA (M=pixels, N=64,
// K=256). Also accumulates per-(b,channel) sum / sumsq for GroupNorm + pooling.
// Block = 256 threads = 8 waves; each wave computes a 16-pixel x 64-channel
// strip: 8 k-steps (K=32 each) x 4 N-tiles = 32 v_wmma per wave.
// ---------------------------------------------------------------------------
__global__ __launch_bounds__(256)
void pass1_reduce_wmma(const float* __restrict__ x,
                       const float* __restrict__ rw,   // (64,256)
                       float* __restrict__ y,
                       float* __restrict__ gsum,
                       float* __restrict__ gsq)
{
    __shared__ __align__(16) float ytile[CREDN][132];  // stride 132: 16B aligned, conflict-free

    const int t    = threadIdx.x;
    const int wave = t >> 5;
    const int lane = t & 31;
    const int bi   = blockIdx.x / (HWP / PBLK);
    const int pblk = (blockIdx.x % (HWP / PBLK)) * PBLK;
    const int p0   = pblk + wave * 16;

    const float* xb = x + (size_t)bi * CIN * HWP;

    const int  row = lane & 15;
    const bool lo  = lane < 16;
    const int  pix = p0 + row;

    v8f acc[4] = {};

    for (int kk = 0; kk < 8; ++kk) {
        const int kbase = kk * 32;
        // A fragment: 16x32 bf16 (rows = pixels, K = input channels)
        v16bf afrag;
#pragma unroll
        for (int j = 0; j < 16; ++j) {
            const int kl = kbase + j + (j < 8 ? 0 : 8) + (lo ? 0 : 8);
            afrag[j] = (__bf16)xb[(size_t)kl * HWP + pix];
        }
#pragma unroll
        for (int n = 0; n < 4; ++n) {
            // B fragment: 32x16 bf16 (rows = K, cols = output channels)
            const float* wr = rw + (size_t)(n * 16 + row) * CIN + kbase + (lo ? 0 : 8);
            v16bf bfrag;
#pragma unroll
            for (int j = 0; j < 16; ++j) bfrag[j] = (__bf16)wr[j + (j < 8 ? 0 : 8)];
            acc[n] = __builtin_amdgcn_wmma_f32_16x16x32_bf16(
                         false, afrag, false, bfrag, (short)0, acc[n], false, false);
        }
    }

    // D layout: VGPR r, lane L -> M = r + (L<16?0:8), N = L%16. Transpose via LDS.
#pragma unroll
    for (int n = 0; n < 4; ++n) {
        const int ch = n * 16 + row;
#pragma unroll
        for (int r = 0; r < 8; ++r) {
            const int pl = wave * 16 + r + (lo ? 0 : 8);
            ytile[ch][pl] = acc[n][r];
        }
    }
    __syncthreads();

    // per-channel partial sums for GN stats + spatial pooling
    if (t < CREDN) {
        float s = 0.f, sq = 0.f;
#pragma unroll 8
        for (int p = 0; p < PBLK; ++p) { const float v = ytile[t][p]; s += v; sq += v * v; }
        atomicAdd(&gsum[bi * CREDN + t], s);
        atomicAdd(&gsq [bi * CREDN + t], sq);
    }

    // coalesced float4 stores of the raw y tile
    float* yb = y + (size_t)bi * CREDN * HWP + pblk;
#pragma unroll
    for (int i = 0; i < 8; ++i) {
        const int idx = t + i * 256;           // 2048 float4's total
        const int c = idx >> 5, q = idx & 31;
        const float4 v = *(const float4*)&ytile[c][q * 4];
        *(float4*)(yb + (size_t)c * HWP + q * 4) = v;
    }
}

// ---------------------------------------------------------------------------
// Pass 2 (one block): fold GN into per-(b,c) affine (alpha,beta), compute
// pooled p, run the 64->16->64 gate MLP, and pre-combine the fuse weights:
// wf[b,c] = fw[c] + gamma[b,c]*fw[64+c].
// ---------------------------------------------------------------------------
__global__ __launch_bounds__(512)
void pass2_stats_gate(const float* __restrict__ gsum, const float* __restrict__ gsq,
                      const float* __restrict__ gn_scale, const float* __restrict__ gn_bias,
                      const float* __restrict__ w1, const float* __restrict__ b1,
                      const float* __restrict__ w2, const float* __restrict__ b2,
                      const float* __restrict__ fw,
                      float* __restrict__ palpha, float* __restrict__ pbeta,
                      float* __restrict__ pwf)
{
    __shared__ float p[BB][CREDN];
    __shared__ float hdn[BB][16];

    const int t  = threadIdx.x;            // 512 = 8 batches * 64 channels
    const int bi = t >> 6;
    const int c  = t & 63;
    const int g  = c >> 3;

    float S = 0.f, Q = 0.f;
#pragma unroll
    for (int j = 0; j < 8; ++j) {
        S += gsum[bi * CREDN + g * 8 + j];
        Q += gsq [bi * CREDN + g * 8 + j];
    }
    const float inv  = 1.f / (8.f * (float)HWP);
    const float m    = S * inv;
    const float var  = Q * inv - m * m;
    const float rstd = rsqrtf(var + 1e-5f);
    const float alpha = gn_scale[c] * rstd;
    const float beta  = gn_bias[c] - m * alpha;
    const float cmean = gsum[bi * CREDN + c] * (1.f / (float)HWP);
    p[bi][c] = cmean * alpha + beta;       // spatial mean of normalized y
    palpha[t] = alpha;
    pbeta[t]  = beta;
    __syncthreads();

    if (t < BB * 16) {
        const int b = t >> 4, h = t & 15;
        float v = b1[h];
        for (int cc = 0; cc < CREDN; ++cc) v += p[b][cc] * w1[h * CREDN + cc];
        hdn[b][h] = fmaxf(v, 0.f);
    }
    __syncthreads();

    float v = b2[c];
#pragma unroll
    for (int h = 0; h < 16; ++h) v += hdn[bi][h] * w2[c * 16 + h];
    const float gamma = 1.f / (1.f + expf(-v));
    pwf[t] = fw[c] + gamma * fw[CREDN + c];
}

// ---------------------------------------------------------------------------
// Pass 3: per 16x16 spatial tile — async-load RAW y tiles (1-px halo) to LDS
// (GLOBAL_LOAD_ASYNC_TO_LDS_B32, ASYNCcnt), fold the GN affine analytically:
// mean kernel sums to 1 and Sobel kernels sum to 0, so with tA=|alpha|:
//   ratio = tA*|y-mu_raw| / (tA*(|gx_raw|+|gy_raw|) + eps)
// SAME zero-padding in normalized space == raw sentinel -beta/alpha.
// Then kappa channel-reduction with wf and out = x*(1 + 0.1*sigmoid(acc)).
// ---------------------------------------------------------------------------
#define CHK 32
__global__ __launch_bounds__(256)
void pass3_fused(const float* __restrict__ y,
                 const float* __restrict__ x,
                 const float* __restrict__ palpha,
                 const float* __restrict__ pbeta,
                 const float* __restrict__ pwf,
                 float* __restrict__ out)
{
    __shared__ float ys[CHK][18 * 18];
    __shared__ float tA[CREDN], padv[CREDN], wfl[CREDN];

    const int t    = threadIdx.x;
    const int bi   = blockIdx.x / 100;
    const int tile = blockIdx.x % 100;
    const int r0   = (tile / 10) * 16;
    const int c0   = (tile % 10) * 16;

    if (t < CREDN) {
        const float a = palpha[bi * CREDN + t];
        const float b = pbeta [bi * CREDN + t];
        tA[t]   = fabsf(a);
        padv[t] = (a != 0.f) ? (-b / a) : 0.f;   // raw value whose normalized value is 0
        wfl[t]  = pwf[bi * CREDN + t];
    }

    const int ty = t >> 4, tx = t & 15;
    const float* yb = y + (size_t)bi * CREDN * HWP;
    float accg = 0.f;

    for (int chunk = 0; chunk < CREDN / CHK; ++chunk) {
        __syncthreads();                        // also covers tA/padv/wfl readiness
        for (int idx = t; idx < CHK * 324; idx += 256) {
            const int c   = idx / 324, pos = idx % 324;
            const int rr  = pos / 18 + r0 - 1;
            const int cc  = pos % 18 + c0 - 1;
            const int cg  = chunk * CHK + c;
            if (rr >= 0 && rr < HH && cc >= 0 && cc < WWD) {
                const float* src = yb + (size_t)cg * HWP + rr * WWD + cc;
#if USE_ASYNC_LDS
                __builtin_amdgcn_global_load_async_to_lds_b32(
                    (g_i32*)(void*)const_cast<float*>(src),
                    (l_i32*)(void*)&ys[c][pos],
                    0, 0);
#else
                ys[c][pos] = *src;
#endif
            } else {
                ys[c][pos] = padv[cg];
            }
        }
#if USE_ASYNC_LDS
        __builtin_amdgcn_s_wait_asynccnt(0);
#endif
        __syncthreads();
#pragma unroll 4
        for (int c = 0; c < CHK; ++c) {
            const int cg = chunk * CHK + c;
            const float* s = &ys[c][ty * 18 + tx];
            const float v00 = s[0],  v01 = s[1],  v02 = s[2];
            const float v10 = s[18], v11 = s[19], v12 = s[20];
            const float v20 = s[36], v21 = s[37], v22 = s[38];
            const float mu = (v00 + v01 + v02 + v10 + v11 + v12 + v20 + v21 + v22) * (1.f / 9.f);
            const float gx = (v00 - v02 + 2.f * (v10 - v12) + v20 - v22) * 0.25f;
            const float gy = (v00 + 2.f * v01 + v02 - v20 - 2.f * v21 - v22) * 0.25f;
            const float tc  = tA[cg];
            const float num = tc * fabsf(v11 - mu);
            const float den = tc * (fabsf(gx) + fabsf(gy)) + 1e-4f;
            const float ratio = fminf(num / den, 2.f);
            accg += wfl[cg] * (1.f - ratio);    // kappa in [-1,1] by construction
        }
    }

    const float a     = 1.f / (1.f + expf(-accg));
    const float scale = 1.f + 0.1f * a;
    const int   pix   = (r0 + ty) * WWD + (c0 + tx);
    const float* xb = x   + (size_t)bi * CIN * HWP + pix;
    float*       ob = out + (size_t)bi * CIN * HWP + pix;
#pragma unroll 4
    for (int cc = 0; cc < CIN; ++cc)
        ob[(size_t)cc * HWP] = xb[(size_t)cc * HWP] * scale;
}

// ---------------------------------------------------------------------------
extern "C" void kernel_launch(void* const* d_in, const int* in_sizes, int n_in,
                              void* d_out, int out_size, void* d_ws, size_t ws_size,
                              hipStream_t stream)
{
    (void)in_sizes; (void)n_in; (void)out_size; (void)ws_size;

    const float* x        = (const float*)d_in[0];
    const float* reduce_w = (const float*)d_in[1];
    const float* gn_scale = (const float*)d_in[2];
    const float* gn_bias  = (const float*)d_in[3];
    const float* gate_w1  = (const float*)d_in[4];
    const float* gate_b1  = (const float*)d_in[5];
    const float* gate_w2  = (const float*)d_in[6];
    const float* gate_b2  = (const float*)d_in[7];
    const float* fuse_w   = (const float*)d_in[8];

    float* ws     = (float*)d_ws;
    float* y      = ws;                    // 13,107,200 floats
    float* gsum   = ws + NYELT;            // 512
    float* gsq    = gsum + BB * CREDN;     // 512
    float* palpha = gsq  + BB * CREDN;     // 512
    float* pbeta  = palpha + BB * CREDN;   // 512
    float* pwf    = pbeta  + BB * CREDN;   // 512

    (void)hipMemsetAsync(gsum, 0, 2 * BB * CREDN * sizeof(float), stream);

    pass1_reduce_wmma<<<NBLK1, 256, 0, stream>>>(x, reduce_w, y, gsum, gsq);
    pass2_stats_gate<<<1, 512, 0, stream>>>(gsum, gsq, gn_scale, gn_bias,
                                            gate_w1, gate_b1, gate_w2, gate_b2,
                                            fuse_w, palpha, pbeta, pwf);
    pass3_fused<<<BB * 100, 256, 0, stream>>>(y, x, palpha, pbeta, pwf, (float*)d_out);
}
